// BVGAE_10325101380093
// MI455X (gfx1250) — matrix operand order repaired and checked by reading, not compiled, not run on hardware
//
#include <hip/hip_runtime.h>
#include <hip/hip_bf16.h>
#include <math.h>

typedef __attribute__((ext_vector_type(16))) _Float16 v16h;
typedef __attribute__((ext_vector_type(8)))  float    v8f;

#define GCN_N    8192
#define GCN_FIN  512
#define GCN_HID  256

// ---------------------------------------------------------------- utilities
__global__ void zero_f32_kernel(float* __restrict__ p, int n) {
    int i = blockIdx.x * blockDim.x + threadIdx.x;
    if (i < n) p[i] = 0.0f;
}

__global__ void deg_kernel(const int* __restrict__ row, float* __restrict__ deg, int E) {
    int e = blockIdx.x * blockDim.x + threadIdx.x;
    if (e < E) atomicAdd(&deg[row[e]], 1.0f);
}

// in-place: deg -> rsqrt(max(deg,1))
__global__ void norm_kernel(float* __restrict__ d, int n) {
    int i = blockIdx.x * blockDim.x + threadIdx.x;
    if (i < n) {
        float v = d[i];
        v = (v < 1.0f) ? 1.0f : v;
        d[i] = rsqrtf(v);
    }
}

// ------------------------------------------------------- WMMA GEMM  Y = (X*norm) @ W^T
// X: (rows, K) row-major f32, scaled per-row by rowScale[m]
// W: (cols, K) row-major f32  (Y[m,n] = sum_k Xs[m,k] * W[n,k])
//
// Block = 256 threads = 8 waves. Each wave owns a 16(M) x 64(N) strip:
// the A fragment is built once per 32-wide K-step and reused across 4 WMMAs,
// quartering A-side traffic and putting 4 v_wmma per inner loop.
// Grid: (rows/128, cols/64).
__global__ void gemm_wmma_nt(const float* __restrict__ X,
                             const float* __restrict__ rowScale,
                             const float* __restrict__ W,
                             float* __restrict__ Y,
                             int K, int ldx, int ldw, int ldy) {
    const int lane  = threadIdx.x & 31;   // wave32
    const int wave  = threadIdx.x >> 5;   // 0..7
    const int l15   = lane & 15;
    const int lhalf = lane >> 4;          // 0 or 1

    const int m     = blockIdx.x * 128 + wave * 16 + l15; // A row owned by this lane
    const int nBase = blockIdx.y * 64;

    const float s = rowScale[m];
    const float* __restrict__ xrow = X + (size_t)m * ldx;

    v8f c[4] = {v8f{}, v8f{}, v8f{}, v8f{}};

    for (int k0 = 0; k0 < K; k0 += 32) {
        // A 16x32 f16 fragment (ISA 7.12.2): lane<16 holds K offsets {0..7,16..23},
        // lane>=16 holds {8..15,24..31}; 2 packed halves per VGPR.
        v16h a;
#pragma unroll
        for (int v = 0; v < 8; ++v) {
            int ka = k0 + ((v < 4) ? 0 : 16) + lhalf * 8 + 2 * (v & 3);
            float2 xv = *(const float2*)(xrow + ka);
            a[2 * v]     = (_Float16)(xv.x * s);
            a[2 * v + 1] = (_Float16)(xv.y * s);
        }
#pragma unroll
        for (int t = 0; t < 4; ++t) {
            // B 32x16 f16 fragment: lane owns column nBase+t*16+l15;
            // lane<16 -> K {0..7,16..23}, lane>=16 -> K {8..15,24..31} (2 halves / VGPR).
            const float* __restrict__ wrow = W + (size_t)(nBase + t * 16 + l15) * ldw;
            v16h b;
#pragma unroll
            for (int v = 0; v < 8; ++v) {
                int kb = k0 + ((v < 4) ? 0 : 16) + lhalf * 8 + 2 * (v & 3);
                float2 wv = *(const float2*)(wrow + kb);
                b[2 * v]     = (_Float16)wv.x;
                b[2 * v + 1] = (_Float16)wv.y;
            }
            c[t] = __builtin_amdgcn_wmma_f32_16x16x32_f16(
                       /*neg_a=*/false, a, /*neg_b=*/false, b,
                       /*c_mod=*/(short)0, c[t], /*reuse_a=*/false, /*reuse_b=*/false);
        }
    }

    // C/D layout: VGPR d -> M = d (lanes 0..15) / d+8 (lanes 16..31), N = lane&15.
#pragma unroll
    for (int t = 0; t < 4; ++t) {
#pragma unroll
        for (int d = 0; d < 8; ++d) {
            int rm = blockIdx.x * 128 + wave * 16 + d + lhalf * 8;
            Y[(size_t)rm * ldy + nBase + t * 16 + l15] = c[t][d];
        }
    }
}

// ---------------------------------------------------------------- SpMM (scatter-add)
// acc[row[e], f] += Y[col[e], f]; one block per edge, 256 feature threads.
// Whole operand set (~20 MB) is resident in the 192 MB L2, so these float
// atomics run at L2 rates.
__global__ void spmm_scatter(const int* __restrict__ row, const int* __restrict__ col,
                             const float* __restrict__ Y, float* __restrict__ acc,
                             int E, int F) {
    int e = blockIdx.x;
    if (e >= E) return;
    int f = threadIdx.x;
    int r = row[e];
    int c = col[e];
    atomicAdd(&acc[(size_t)r * F + f], Y[(size_t)c * F + f]);
}

// buf[i,f] = (relu?) max(0, norm[i]*buf[i,f]) : norm[i]*buf[i,f]
__global__ void post_scale(float* __restrict__ buf, const float* __restrict__ norm,
                           int F, int do_relu) {
    int node = blockIdx.x;
    int f = threadIdx.x;
    size_t idx = (size_t)node * F + f;
    float v = buf[idx] * norm[node];
    if (do_relu) v = fmaxf(v, 0.0f);
    buf[idx] = v;
}

// ------------------------------------------------- alpha/beta heads (dot + ELU + 1.5)
// One wave per node: coalesced lane-strided reads + wave32 butterfly reduction.
__global__ void ab_kernel(const float* __restrict__ h1,
                          const float* __restrict__ wa, const float* __restrict__ ba,
                          const float* __restrict__ wb, const float* __restrict__ bb,
                          float* __restrict__ alpha, float* __restrict__ beta,
                          int N, int F) {
    const int lane = threadIdx.x & 31;
    const int wave = threadIdx.x >> 5;
    const int n = blockIdx.x * 8 + wave;
    if (n >= N) return;
    const float* __restrict__ r = h1 + (size_t)n * F;
    float da = 0.0f, db = 0.0f;
    for (int f = lane; f < F; f += 32) {
        float v = r[f];
        da = fmaf(v, wa[f], da);
        db = fmaf(v, wb[f], db);
    }
#pragma unroll
    for (int off = 16; off > 0; off >>= 1) {
        da += __shfl_xor(da, off, 32);
        db += __shfl_xor(db, off, 32);
    }
    if (lane == 0) {
        da += ba[0];
        db += bb[0];
        da = (da > 0.0f) ? da : (expf(da) - 1.0f);
        db = (db > 0.0f) ? db : (expf(db) - 1.0f);
        alpha[n] = da + 1.5f;
        beta[n]  = db + 1.5f;
    }
}

// ------------------------------------- final 512MB writer: out = alpha_p ++ beta_p
// grid: (N/1024, N); each thread writes one float4 of alpha_p row i and one of
// beta_p. This kernel is the roofline limiter (~512 MB stores @ 23.3 TB/s).
__global__ void outer_kernel(const float* __restrict__ alpha,
                             const float* __restrict__ beta,
                             float* __restrict__ out, int N) {
    int i  = blockIdx.y;
    int j0 = (blockIdx.x * blockDim.x + threadIdx.x) * 4;
    float ai = alpha[i];
    float bi = beta[i];
    float4 a4 = *(const float4*)(alpha + j0);
    float4 b4 = *(const float4*)(beta + j0);
    size_t base = (size_t)i * N + j0;
    float4 oa = make_float4(ai + a4.x, ai + a4.y, ai + a4.z, ai + a4.w);
    float4 ob = make_float4(bi + b4.x, bi + b4.y, bi + b4.z, bi + b4.w);
    *(float4*)(out + base) = oa;
    *(float4*)(out + (size_t)N * N + base) = ob;
}

// ------------------------------------------------------------------- launch
extern "C" void kernel_launch(void* const* d_in, const int* in_sizes, int n_in,
                              void* d_out, int out_size, void* d_ws, size_t ws_size,
                              hipStream_t stream) {
    const int*   row = (const int*)  d_in[0];
    const int*   col = (const int*)  d_in[1];
    const float* h   = (const float*)d_in[2];
    const float* W0  = (const float*)d_in[3];
    const float* W1  = (const float*)d_in[4];
    const float* wa  = (const float*)d_in[5];
    const float* ba  = (const float*)d_in[6];
    const float* wb  = (const float*)d_in[7];
    const float* bb  = (const float*)d_in[8];
    float* out = (float*)d_out;

    const int N = GCN_N, FIN = GCN_FIN, HID = GCN_HID;
    const int E = in_sizes[0];

    // workspace layout (floats): norm[N] | B1[N*HID] | B2[N*HID] | alpha[N] | beta[N]
    float* norm  = (float*)d_ws;
    float* B1    = norm + N;
    float* B2    = B1 + (size_t)N * HID;
    float* alpha = B2 + (size_t)N * HID;
    float* beta  = alpha + N;

    // degree -> norm
    zero_f32_kernel<<<(N + 255) / 256, 256, 0, stream>>>(norm, N);
    deg_kernel<<<(E + 255) / 256, 256, 0, stream>>>(row, norm, E);
    norm_kernel<<<(N + 255) / 256, 256, 0, stream>>>(norm, N);

    dim3 gTiles(N / 128, HID / 64);  // 64 x 4 blocks, 8 waves each

    // layer 0: Y0 = (h*norm) @ W0^T  -> B1
    gemm_wmma_nt<<<gTiles, 256, 0, stream>>>(h, norm, W0, B1, FIN, FIN, FIN, HID);
    // acc0 = A @ Y0 -> B2 ; h0 = relu(norm*acc0) in place
    zero_f32_kernel<<<(N * HID + 255) / 256, 256, 0, stream>>>(B2, N * HID);
    spmm_scatter<<<E, HID, 0, stream>>>(row, col, B1, B2, E, HID);
    post_scale<<<N, HID, 0, stream>>>(B2, norm, HID, 1);

    // layer 1: Y1 = (h0*norm) @ W1^T -> B1
    gemm_wmma_nt<<<gTiles, 256, 0, stream>>>(B2, norm, W1, B1, HID, HID, HID, HID);
    // acc1 = A @ Y1 -> B2 ; h1 = norm*acc1 in place
    zero_f32_kernel<<<(N * HID + 255) / 256, 256, 0, stream>>>(B2, N * HID);
    spmm_scatter<<<E, HID, 0, stream>>>(row, col, B1, B2, E, HID);
    post_scale<<<N, HID, 0, stream>>>(B2, norm, HID, 0);

    // heads: one wave per node
    ab_kernel<<<N / 8, 256, 0, stream>>>(B2, wa, ba, wb, bb, alpha, beta, N, HID);

    // 512 MB broadcast-sum output (store-bound roofline: ~22 us at 23.3 TB/s)
    dim3 gOut(N / (256 * 4), N);
    outer_kernel<<<gOut, 256, 0, stream>>>(alpha, beta, out, N);
}
